// Linear_41609643163832
// MI455X (gfx1250) — 8-bit matrix operand order repaired and checked by reading, compiled, not run on hardware
//
#include <hip/hip_runtime.h>


typedef __attribute__((ext_vector_type(16))) int   v16i;
typedef __attribute__((ext_vector_type(8)))  float v8f;

#define KDIM   2048
#define NDIM   2048
#define KB     (KDIM / 128)
#define FP8MAX 448.0f

__device__ __forceinline__ void async_copy_b128(void* lds, const void* gaddr) {
    asm volatile("global_load_async_to_lds_b128 %0, %1, off"
                 :: "v"((__attribute__((address_space(3))) void*)lds),
                    "v"(gaddr)
                 : "memory");
}
__device__ __forceinline__ void async_copy_b32(void* lds, const void* gaddr) {
    asm volatile("global_load_async_to_lds_b32 %0, %1, off"
                 :: "v"((__attribute__((address_space(3))) void*)lds),
                    "v"(gaddr)
                 : "memory");
}
__device__ __forceinline__ void wait_async0() {
    asm volatile("s_wait_asynccnt 0x0" ::: "memory");
}

__device__ __forceinline__ unsigned int f2e4m3(float x) {
    unsigned int b    = __float_as_uint(x);
    unsigned int sign = (b >> 24) & 0x80u;
    unsigned int ax   = b & 0x7FFFFFFFu;
    if (ax >= 0x7F800000u) return sign | 0x7Fu;
    if (ax >= 0x43E00000u) return sign | 0x7Eu;
    if (ax <  0x3C800000u) {
        float f = __uint_as_float(ax);
        unsigned int m = (unsigned int)rintf(f * 512.0f);
        return sign | m;
    }
    unsigned int mant = ax & 0x7FFFFFu;
    unsigned int keep = mant >> 20;
    unsigned int rem  = mant & 0xFFFFFu;
    unsigned int rnd  = (rem > 0x80000u) || (rem == 0x80000u && (keep & 1u));
    unsigned int e8   = (ax >> 23) - 120u;
    return sign | (((e8 << 3) | keep) + rnd);
}

__device__ __forceinline__ unsigned int pack4_e4m3(float4 v) {
    return  f2e4m3(v.x)
         | (f2e4m3(v.y) << 8)
         | (f2e4m3(v.z) << 16)
         | (f2e4m3(v.w) << 24);
}

__global__ __launch_bounds__(256)
void quant_x_kernel(const float* __restrict__ x,
                    unsigned char* __restrict__ xq,
                    float* __restrict__ xs, int M) {
    const int wave = blockIdx.x * (blockDim.x >> 5) + (threadIdx.x >> 5);
    const int lane = threadIdx.x & 31;
    const int row  = wave / KB;
    const int kb   = wave % KB;
    if (row >= M) return;

    const float4 v = *(const float4*)(x + (size_t)row * KDIM + (size_t)kb * 128 + lane * 4);
    float a = fmaxf(fmaxf(fabsf(v.x), fabsf(v.y)), fmaxf(fabsf(v.z), fabsf(v.w)));
    #pragma unroll
    for (int m = 16; m > 0; m >>= 1)
        a = fmaxf(a, __shfl_xor(a, m, 32));

    float s = a / FP8MAX;
    s = (s == 0.0f) ? 1.0f : s;
    const float inv = 1.0f / s;
    const unsigned int pk =
        pack4_e4m3(make_float4(v.x * inv, v.y * inv, v.z * inv, v.w * inv));

    const int tile = row >> 4, mm = row & 15;
    const int half = (lane >> 1) & 1;
    const int fv   = ((lane >> 4) << 3) + (((lane >> 2) & 3) << 1) + (lane & 1);
    ((unsigned int*)xq)[((size_t)(tile * KB + kb) * 32 + half * 16 + mm) * 16 + fv] = pk;
    if (lane == 0) xs[(size_t)row * KB + kb] = s;
}

__global__ __launch_bounds__(256)
void quant_w_kernel(const float* __restrict__ w, unsigned char* __restrict__ wq) {
    const size_t i = (size_t)blockIdx.x * blockDim.x + threadIdx.x;
    const int n = (int)(i >> 9);
    const int d = (int)(i & 511);
    const float4 v = *(const float4*)(w + ((size_t)n << 11) + (d << 2));
    const unsigned int pk = pack4_e4m3(v);

    const int kb   = d >> 5;
    const int L    = d & 31;
    const int half = (L >> 1) & 1;
    const int fv   = ((L >> 4) << 3) + (((L >> 2) & 3) << 1) + (L & 1);
    ((unsigned int*)wq)[((size_t)((n >> 4) * KB + kb) * 32 + half * 16 + (n & 15)) * 16 + fv] = pk;
}

__global__ __launch_bounds__(256)
void gemm_fp8_kernel(const unsigned char* __restrict__ xq,
                     const float*         __restrict__ xs,
                     const unsigned char* __restrict__ wq,
                     const float*         __restrict__ wscale,
                     float*               __restrict__ out, int M) {
    __shared__ int   lA[2][4096];
    __shared__ int   lB[2][4096];
    __shared__ float lSx[2][128];

    const int tid  = threadIdx.x;
    const int lane = tid & 31;
    const int wid  = tid >> 5;
    const int m0   = blockIdx.y * 128;
    const int n0   = blockIdx.x * 128;
    const int nb   = n0 >> 7;

    const int mw = wid >> 1;
    const int nw = wid & 1;

    v8f acc[2][4];
    #pragma unroll
    for (int i = 0; i < 2; ++i)
        #pragma unroll
        for (int j = 0; j < 4; ++j)
            acc[i][j] = (v8f){0.f, 0.f, 0.f, 0.f, 0.f, 0.f, 0.f, 0.f};

    auto stage = [&](int buf, int kbs) {
        #pragma unroll
        for (int r = 0; r < 4; ++r) {
            const int c     = tid + r * 256;
            const int t     = c >> 7;
            const int inner = (c & 127) << 4;
            async_copy_b128((char*)&lA[buf][0] + t * 2048 + inner,
                            xq + ((size_t)((m0 >> 4) + t) * KB + kbs) * 2048 + inner);
            async_copy_b128((char*)&lB[buf][0] + t * 2048 + inner,
                            wq + ((size_t)((n0 >> 4) + t) * KB + kbs) * 2048 + inner);
        }
        if (wid < 4) {
            const int i = wid * 32 + lane;
            async_copy_b32(&lSx[buf][i], xs + (size_t)(m0 + i) * KB + kbs);
        }
    };

    stage(0, 0);
    wait_async0();
    __syncthreads();

    for (int kb = 0; kb < KB; ++kb) {
        const int cur = kb & 1;
        if (kb + 1 < KB) stage(cur ^ 1, kb + 1);

        const float sw = wscale[nb * KB + kb];

        union Frag { v16i v; uint4 q[4]; } af[2], bf[4];
        #pragma unroll
        for (int i = 0; i < 2; ++i) {
            const uint4* p = (const uint4*)&lA[cur][((mw * 2 + i) * 32 + lane) * 16];
            af[i].q[0] = p[0]; af[i].q[1] = p[1]; af[i].q[2] = p[2]; af[i].q[3] = p[3];
        }
        #pragma unroll
        for (int j = 0; j < 4; ++j) {
            const uint4* p = (const uint4*)&lB[cur][((nw * 4 + j) * 32 + lane) * 16];
            bf[j].q[0] = p[0]; bf[j].q[1] = p[1]; bf[j].q[2] = p[2]; bf[j].q[3] = p[3];
        }

        const int half = lane >> 4;
        float4 sxlo[2], sxhi[2];
        #pragma unroll
        for (int i = 0; i < 2; ++i) {
            const float4* sp = (const float4*)&lSx[cur][(mw * 2 + i) * 16 + half * 8];
            sxlo[i] = sp[0];
            sxhi[i] = sp[1];
        }

        #pragma unroll
        for (int i = 0; i < 2; ++i) {
            #pragma unroll
            for (int j = 0; j < 4; ++j) {
                v8f z = (v8f){0.f, 0.f, 0.f, 0.f, 0.f, 0.f, 0.f, 0.f};
                v8f p = __builtin_amdgcn_wmma_f32_16x16x128_fp8_fp8(
                            af[i].v, bf[j].v, (short)0, z, false, false);
                acc[i][j][0] += (sw * sxlo[i].x) * p[0];
                acc[i][j][1] += (sw * sxlo[i].y) * p[1];
                acc[i][j][2] += (sw * sxlo[i].z) * p[2];
                acc[i][j][3] += (sw * sxlo[i].w) * p[3];
                acc[i][j][4] += (sw * sxhi[i].x) * p[4];
                acc[i][j][5] += (sw * sxhi[i].y) * p[5];
                acc[i][j][6] += (sw * sxhi[i].z) * p[6];
                acc[i][j][7] += (sw * sxhi[i].w) * p[7];
            }
        }

        wait_async0();
        __syncthreads();
    }

    const int half = lane >> 4;
    const int nn   = lane & 15;
    #pragma unroll
    for (int i = 0; i < 2; ++i) {
        #pragma unroll
        for (int j = 0; j < 4; ++j) {
            const int tm = mw * 2 + i, tn = nw * 4 + j;
            float* o = out + (size_t)(m0 + tm * 16 + half * 8) * NDIM
                           + n0 + tn * 16 + nn;
            #pragma unroll
            for (int v = 0; v < 8; ++v)
                o[(size_t)v * NDIM] = acc[i][j][v];
        }
    }
}

extern "C" void kernel_launch(void* const* d_in, const int* in_sizes, int n_in,
                              void* d_out, int out_size, void* d_ws, size_t ws_size,
                              hipStream_t stream) {
    (void)n_in; (void)out_size; (void)ws_size;

    const float* x       = (const float*)d_in[0];
    const float* w_fp8   = (const float*)d_in[1];
    const float* w_scale = (const float*)d_in[2];
    float*       out     = (float*)d_out;

    const int M = in_sizes[0] / KDIM;

    unsigned char* xq       = (unsigned char*)d_ws;
    const size_t   xq_bytes = (size_t)M * KDIM;
    float*         xsc      = (float*)((char*)d_ws + xq_bytes);
    const size_t   xs_bytes = (size_t)M * KB * sizeof(float);
    unsigned char* wqp      = (unsigned char*)((char*)d_ws + xq_bytes + xs_bytes);

    const int qx_blocks = (M * KB) / 8;
    quant_x_kernel<<<qx_blocks, 256, 0, stream>>>(x, xq, xsc, M);

    quant_w_kernel<<<(NDIM * KDIM / 4) / 256, 256, 0, stream>>>(w_fp8, wqp);

    dim3 grid(NDIM / 128, M / 128);
    gemm_fp8_kernel<<<grid, 256, 0, stream>>>(xq, xsc, wqp, w_scale, out, M);
}
